// TimeMix_40106404610589
// MI455X (gfx1250) — matrix operand order repaired and checked by reading, not compiled, not run on hardware
//
#include <hip/hip_runtime.h>
#include <hip/hip_bf16.h>

// ---------------------------------------------------------------------------
// RWKV-6 TimeMix forward for MI455X (gfx1250, wave32, WMMA).
// Big GEMMs use v_wmma_f32_16x16x32_bf16. Weights (B) are staged into a
// double-buffered LDS tile via GLOBAL_LOAD_ASYNC_TO_LDS_B128 (ASYNCcnt path)
// when the toolchain exposes the builtin, else via reg staging + ds_store.
// Activations (A) stream from global with a one-tile-ahead register prefetch.
// The chunked recurrence stays in f32 VALU for numerical fidelity.
// ---------------------------------------------------------------------------

typedef __attribute__((ext_vector_type(16))) __bf16 v16bf;
typedef __attribute__((ext_vector_type(8)))  __bf16 v8bf;
typedef __attribute__((ext_vector_type(8)))  float  v8f;

#if defined(__has_builtin)
#if __has_builtin(__builtin_amdgcn_global_load_async_to_lds_b128)
#define ASYNC_LDS 1
#endif
#endif
#ifndef ASYNC_LDS
#define ASYNC_LDS 0
#endif

#if ASYNC_LDS
// Parameter types per hipcc diagnostic: pointer to int vector_size(16),
// AS(1) for the global source, AS(3) for the LDS destination.
typedef int v4i_raw __attribute__((vector_size(4 * sizeof(int))));
typedef __attribute__((address_space(1))) v4i_raw as1_v4i;
typedef __attribute__((address_space(3))) v4i_raw as3_v4i;
#if defined(__has_builtin) && __has_builtin(__builtin_amdgcn_s_wait_asynccnt)
#define WAIT_ASYNC0() __builtin_amdgcn_s_wait_asynccnt(0)
#else
#define WAIT_ASYNC0() asm volatile("s_wait_asynccnt 0x0" ::: "memory")
#endif
#endif

__device__ __forceinline__ __bf16 f2bf(float f) {
  unsigned u = __builtin_bit_cast(unsigned, f);
  unsigned r = u + 0x7FFFu + ((u >> 16) & 1u);   // round-to-nearest-even
  unsigned short h = (unsigned short)(r >> 16);
  return __builtin_bit_cast(__bf16, h);
}

// ---------------------------------------------------------------------------
// Conversion kernels
// ---------------------------------------------------------------------------
__global__ __launch_bounds__(256) void cvt_bf16(const float* __restrict__ s,
                                                __bf16* __restrict__ d, int n) {
  int i = blockIdx.x * 256 + threadIdx.x;
  if (i < n) d[i] = f2bf(s[i]);
}

// src (rows x cols) row-major -> dst (cols x rows) row-major, bf16
__global__ __launch_bounds__(256) void cvt_bf16_t(const float* __restrict__ s,
                                                  __bf16* __restrict__ d,
                                                  int rows, int cols) {
  int i = blockIdx.x * 256 + threadIdx.x;
  if (i < rows * cols) {
    int r = i / cols;
    int c = i - r * cols;
    d[(size_t)c * rows + r] = f2bf(s[i]);
  }
}

// ---------------------------------------------------------------------------
// premix: dx = xprev - x ; xmix = bf16(x + dx * time_maa_x)
// ---------------------------------------------------------------------------
__global__ __launch_bounds__(256)
void premix(const float* __restrict__ x, const float* __restrict__ shift,
            const float* __restrict__ tmx, float* __restrict__ dx,
            __bf16* __restrict__ xmix, int T, int C) {
  int i = blockIdx.x * 256 + threadIdx.x;
  int c  = i % C;
  int tc = i % (T * C);
  int b  = i / (T * C);
  float xp = (tc < C) ? shift[b * C + c] : x[i - C];
  float xv = x[i];
  float d  = xp - xv;
  dx[i]   = d;
  xmix[i] = f2bf(xv + d * tmx[c]);
}

// ---------------------------------------------------------------------------
// WMMA bf16 GEMM:  C[M,N] = A[M,K] @ Bw[N,K]^T   (f32 accum, v_wmma 16x16x32)
//
// Block = 256 threads (8 waves), tile 128(M) x 64(N), K-step 64.
// B tile (64x64 bf16, 8KB) double-buffered in LDS, one barrier per K-tile.
// Async path: tile kt+1 issued straight into LDS after compute(kt) (all
// readers of that buffer passed barrier(kt)); s_wait_asynccnt 0 + barrier at
// top of iter kt+1 publish it. Fragment ds_loads are software-pipelined (load
// fragment u+1 while WMMA consumes u). LDS rows padded to 72 elements: the
// 16-lane b128 fragment reads hit a perfect stride-4 bank permutation.
//
// A fragment (ISA 7.12.2, 16-bit A 16x32): lane(m=l16, half): K = half*8+{0..7},
// 16+half*8+{0..7}. B fragment (32x16): same K order as A: half*8+{0..7},
// 16+half*8+{0..7}, column n = l16. C/D: lane n = l16; VGPR r holds m = r + half*8.
// MODE: 0 = f32 row-major, 1 = tanh f32 row-major, 2 = f32 scatter (B,H,T,64),
//       3 = silu f32 row-major
// ---------------------------------------------------------------------------
#define LPITCH 72
template <int MODE>
__global__ __launch_bounds__(256, 1)
void gemm_bf16(const __bf16* __restrict__ A, const __bf16* __restrict__ Bw,
               float* __restrict__ Cout, int M, int N, int Kd, int Hh, int Tt) {
  __shared__ __bf16 Bbuf[2][64 * LPITCH];

  const int tid  = threadIdx.x;
  const int lane = tid & 31;
  const int wv   = tid >> 5;
  const int hlf  = lane >> 4;
  const int l16  = lane & 15;
  const int m0   = blockIdx.y * 128 + wv * 16;
  const int n0   = blockIdx.x * 64;

  // B staging: 64x64 bf16 tile = 512 uint4 chunks, 2 per thread.
  int brow[2], bq[2], bgr[2];
#pragma unroll
  for (int i = 0; i < 2; ++i) {
    int c = tid + i * 256;
    brow[i] = c >> 3;
    bq[i]   = (c & 7) * 8;
    int gr = n0 + brow[i];
    bgr[i] = (gr < N) ? gr : (N - 1);  // clamp ragged N; stores are guarded
  }

#if ASYNC_LDS
  auto asyncB = [&](int k0, int buf) {
#pragma unroll
    for (int i = 0; i < 2; ++i) {
      const __bf16* gp = Bw + (size_t)bgr[i] * Kd + k0 + bq[i];
      __bf16* lp = &Bbuf[buf][brow[i] * LPITCH + bq[i]];
      __builtin_amdgcn_global_load_async_to_lds_b128((as1_v4i*)gp,
                                                     (as3_v4i*)lp, 0, 0);
    }
  };
#else
  uint4 breg[2];
  auto loadB = [&](int k0) {
#pragma unroll
    for (int i = 0; i < 2; ++i)
      breg[i] = *reinterpret_cast<const uint4*>(
          Bw + (size_t)bgr[i] * Kd + k0 + bq[i]);
  };
  auto storeB = [&](int buf) {
#pragma unroll
    for (int i = 0; i < 2; ++i)
      *reinterpret_cast<uint4*>(&Bbuf[buf][brow[i] * LPITCH + bq[i]]) = breg[i];
  };
#endif

  // A stream: wave-private rows; raw 16B chunks {s0.lo, s0.hi, s1.lo, s1.hi},
  // assembled into fragments only at use time (pure register ops).
  const __bf16* pa = A + (size_t)(m0 + l16) * Kd + hlf * 8;
  uint4 aC[4], aN[4];
  auto loadA = [&](int k0, uint4* dst) {
    dst[0] = *reinterpret_cast<const uint4*>(pa + k0);
    dst[1] = *reinterpret_cast<const uint4*>(pa + k0 + 16);
    dst[2] = *reinterpret_cast<const uint4*>(pa + k0 + 32);
    dst[3] = *reinterpret_cast<const uint4*>(pa + k0 + 48);
  };
  auto ldBfrag = [&](int buf, int j, int s) -> v16bf {
    const __bf16* pbr = &Bbuf[buf][(j * 16 + l16) * LPITCH + s * 32 + hlf * 8];
    v8bf blo = *reinterpret_cast<const v8bf*>(pbr);
    v8bf bhi = *reinterpret_cast<const v8bf*>(pbr + 16);
    return __builtin_shufflevector(blo, bhi, 0, 1, 2, 3, 4, 5, 6, 7, 8, 9, 10,
                                   11, 12, 13, 14, 15);
  };

  v8f acc[4] = {v8f{}, v8f{}, v8f{}, v8f{}};
  const int KT = Kd >> 6;  // K tiles of 64 (K is always a multiple of 64 here)

#if ASYNC_LDS
  asyncB(0, 0);
#else
  loadB(0);
  storeB(0);
#endif
  loadA(0, aC);

  for (int kt = 0; kt < KT; ++kt) {
    const int cur = kt & 1;
    if (kt + 1 < KT) {
#if !ASYNC_LDS
      loadB((kt + 1) << 6);      // overlaps the WMMA block below
#endif
      loadA((kt + 1) << 6, aN);
    }
    if (kt + 2 < KT) {           // GL2 prefetch two tiles ahead
      __builtin_prefetch(pa + ((kt + 2) << 6), 0, 1);
      __builtin_prefetch(Bw + (size_t)bgr[0] * Kd + ((kt + 2) << 6) + bq[0], 0, 1);
    }
#if ASYNC_LDS
    WAIT_ASYNC0();    // tile kt's async LDS writes complete (this wave)
#endif
    __syncthreads();  // ... and published; prev readers of Bbuf[!cur] done

    v16bf afr0, afr1;
    {
      v8bf l0 = __builtin_bit_cast(v8bf, aC[0]);
      v8bf h0 = __builtin_bit_cast(v8bf, aC[1]);
      afr0 = __builtin_shufflevector(l0, h0, 0, 1, 2, 3, 4, 5, 6, 7, 8, 9, 10,
                                     11, 12, 13, 14, 15);
      v8bf l1 = __builtin_bit_cast(v8bf, aC[2]);
      v8bf h1 = __builtin_bit_cast(v8bf, aC[3]);
      afr1 = __builtin_shufflevector(l1, h1, 0, 1, 2, 3, 4, 5, 6, 7, 8, 9, 10,
                                     11, 12, 13, 14, 15);
    }
    v16bf bfr[2];
    bfr[0] = ldBfrag(cur, 0, 0);
#pragma unroll
    for (int u = 0; u < 8; ++u) {       // u = s*4 + j; pipeline frag u+1
      const int s = u >> 2, j = u & 3;
      if (u < 7) bfr[(u + 1) & 1] = ldBfrag(cur, (u + 1) & 3, (u + 1) >> 2);
      acc[j] = __builtin_amdgcn_wmma_f32_16x16x32_bf16(
          false, (s == 0) ? afr0 : afr1, false, bfr[u & 1], (short)0, acc[j],
          false, false);
    }

    if (kt + 1 < KT) {
#if ASYNC_LDS
      asyncB((kt + 1) << 6, (kt + 1) & 1);  // safe: all passed barrier(kt)
#else
      storeB((kt + 1) & 1);
#endif
#pragma unroll
      for (int i = 0; i < 4; ++i) aC[i] = aN[i];
    }
  }

#pragma unroll
  for (int j = 0; j < 4; ++j) {
    int n = n0 + j * 16 + l16;
    if (n >= N) continue;
#pragma unroll
    for (int r = 0; r < 8; ++r) {
      int m = m0 + r + hlf * 8;
      float v = acc[j][r];
      if (MODE == 1) v = tanhf(v);
      if (MODE == 3) v = v / (1.f + __expf(-v));
      if (MODE == 2) {
        int b  = m / Tt;
        int t  = m - b * Tt;
        int h  = n >> 6;
        int kk = n & 63;
        Cout[(((size_t)b * Hh + h) * Tt + t) * 64 + kk] = v;
      } else {
        Cout[(size_t)m * N + n] = v;
      }
    }
  }
}

// ---------------------------------------------------------------------------
// mix_lerp: m = einsum(xxx(BT,5,32), w2(5,32,C)) fused with the 5 lerps.
// ---------------------------------------------------------------------------
__global__ __launch_bounds__(256)
void mix_lerp(const float* __restrict__ x, const float* __restrict__ dx,
              const float* __restrict__ xxx, const float* __restrict__ w2,
              const float* __restrict__ tmw, const float* __restrict__ tmk,
              const float* __restrict__ tmv, const float* __restrict__ tmr,
              const float* __restrict__ tmg, __bf16* __restrict__ xk,
              __bf16* __restrict__ xv, __bf16* __restrict__ xr,
              __bf16* __restrict__ xg, __bf16* __restrict__ xw, int C) {
  __shared__ float xs[160];
  int bt = blockIdx.x;
  for (int i = threadIdx.x; i < 160; i += 256) xs[i] = xxx[(size_t)bt * 160 + i];
  __syncthreads();
  for (int c = threadIdx.x; c < C; c += 256) {
    float mm[5];
#pragma unroll
    for (int f = 0; f < 5; ++f) {
      float s = 0.f;
#pragma unroll 8
      for (int e = 0; e < 32; ++e)
        s += xs[f * 32 + e] * w2[((size_t)(f * 32 + e)) * C + c];
      mm[f] = s;
    }
    size_t idx = (size_t)bt * C + c;
    float xv_ = x[idx], d = dx[idx];
    xw[idx] = f2bf(xv_ + d * (tmw[c] + mm[0]));
    xk[idx] = f2bf(xv_ + d * (tmk[c] + mm[1]));
    xv[idx] = f2bf(xv_ + d * (tmv[c] + mm[2]));
    xr[idx] = f2bf(xv_ + d * (tmr[c] + mm[3]));
    xg[idx] = f2bf(xv_ + d * (tmg[c] + mm[4]));
  }
}

// ---------------------------------------------------------------------------
// decay: wlog[b,h,t,k] = max(-exp(td[c] + wt[bt]@td_w2[:,c]), log(0.005))
// ---------------------------------------------------------------------------
__global__ __launch_bounds__(256)
void decay(const float* __restrict__ wt, const float* __restrict__ tdw2,
           const float* __restrict__ td, float* __restrict__ wlog, int C,
           int T, int Hh) {
  __shared__ float wsv[64];
  int bt = blockIdx.x;
  for (int i = threadIdx.x; i < 64; i += 256) wsv[i] = wt[(size_t)bt * 64 + i];
  __syncthreads();
  const float LMIN = -5.2983174f;  // log(0.005)
  int b = bt / T, t = bt - (bt / T) * T;
  for (int c = threadIdx.x; c < C; c += 256) {
    float s = td[c];
#pragma unroll 8
    for (int e = 0; e < 64; ++e) s += wsv[e] * tdw2[(size_t)e * C + c];
    float lw = -__expf(s);
    if (lw < LMIN) lw = LMIN;
    int h = c >> 6, kk = c & 63;
    wlog[(((size_t)b * Hh + h) * T + t) * 64 + kk] = lw;
  }
}

// ---------------------------------------------------------------------------
// chunk kernel: per (b,h,n) chunk of 32 tokens x 64 channels.
// ---------------------------------------------------------------------------
#define CT 32
#define CK 64
__global__ __launch_bounds__(256)
void chunk_kernel(const float* __restrict__ rbuf, const float* __restrict__ kbuf,
                  const float* __restrict__ vbuf, const float* __restrict__ wlog,
                  const float* __restrict__ u, float* __restrict__ outp,
                  float* __restrict__ wkvb, float* __restrict__ wsb,
                  float* __restrict__ rwi, int Hh, int T, int N) {
  __shared__ float cum[CT][CK];
  __shared__ float rd[CT][CK];
  __shared__ float kd[CT][CK];
  __shared__ float vs[CT][CK];
  __shared__ float kwi[CT][CK];
  __shared__ float amat[CT][CT];
  __shared__ float offv[CK];
  __shared__ float wslog[CK];
  __shared__ float diag[CT];

  int cid = blockIdx.x;
  int n = cid % N;
  int bh = cid / N;
  int h = bh % Hh;
  size_t base = ((size_t)bh * T + (size_t)n * CT) * CK;
  int tid = threadIdx.x;

  if (tid < CK) {  // sequential cumsum over t per channel
    float s = 0.f;
    for (int t = 0; t < CT; ++t) {
      s += wlog[base + (size_t)t * CK + tid];
      cum[t][tid] = s;
    }
    wslog[tid] = s;
    offv[tid] = cum[CT / 2 - 1][tid];  // shifted[T/2] = cum[15]
    wsb[(size_t)cid * CK + tid] = __expf(s);
  }
  if (tid >= 64 && tid < 96) {  // diag[t] = sum_k r*u*k
    int t = tid - 64;
    float s = 0.f;
    const float* ur = u + h * CK;
    for (int kk = 0; kk < CK; ++kk)
      s += rbuf[base + (size_t)t * CK + kk] * ur[kk] *
           kbuf[base + (size_t)t * CK + kk];
    diag[t] = s;
  }
  __syncthreads();

  for (int i = tid; i < CT * CK; i += 256) {
    int t = i >> 6, kk = i & 63;
    float wl = wlog[base + i];
    float cc = cum[t][kk];
    float shifted = cc - wl;  // == cum[t-1] (0 at t=0)
    float off = offv[kk];
    float rv = rbuf[base + i];
    float kv = kbuf[base + i];
    vs[t][kk] = vbuf[base + i];
    rd[t][kk] = rv * __expf(shifted - off);       // r * r_decay
    kd[t][kk] = kv * __expf(off - cc);            // k * k_inv_decay
    kwi[t][kk] = kv * __expf(wslog[kk] - cc);     // k * w_inter
    rwi[base + i] = rv * __expf(shifted);         // r * w_intra
  }
  __syncthreads();

  for (int i = tid; i < CT * CT; i += 256) {  // a = tril(rd @ kd^T, -1)
    int t = i >> 5, s2 = i & 31;
    float acc = 0.f;
    if (s2 < t) {
#pragma unroll 8
      for (int kk = 0; kk < CK; ++kk) acc += rd[t][kk] * kd[s2][kk];
    }
    amat[t][s2] = acc;
  }
  __syncthreads();

  for (int i = tid; i < CT * CK; i += 256) {  // out_intra = a @ v + diag*v
    int t = i >> 6, kk = i & 63;
    float acc = diag[t] * vs[t][kk];
    for (int s2 = 0; s2 < t; ++s2) acc += amat[t][s2] * vs[s2][kk];
    outp[base + i] = acc;
  }

  for (int i = tid; i < CK * CK; i += 256) {  // wkv[k][v] = sum_t kwi*v
    int kk = i >> 6, kv = i & 63;
    float acc = 0.f;
#pragma unroll 8
    for (int t = 0; t < CT; ++t) acc += kwi[t][kk] * vs[t][kv];
    wkvb[(size_t)cid * CK * CK + i] = acc;
  }
}

// ---------------------------------------------------------------------------
// scan kernel: one block per (b,h); sequential over N chunks.
// ---------------------------------------------------------------------------
__global__ __launch_bounds__(256)
void scan_kernel(const float* __restrict__ state_in, const float* __restrict__ wsb,
                 const float* __restrict__ wkvb, const float* __restrict__ rwi,
                 float* __restrict__ outp, float* __restrict__ state_out, int T,
                 int N) {
  __shared__ float st[CK][CK];
  __shared__ float rs[CT][CK];
  __shared__ float wsv[CK];
  int bh = blockIdx.x;
  for (int i = threadIdx.x; i < CK * CK; i += 256)
    st[i >> 6][i & 63] = state_in[(size_t)bh * CK * CK + i];
  __syncthreads();

  for (int n = 0; n < N; ++n) {
    size_t cb = (size_t)bh * N + n;
    size_t tbase = ((size_t)bh * T + (size_t)n * CT) * CK;
    for (int i = threadIdx.x; i < CT * CK; i += 256)
      rs[i >> 6][i & 63] = rwi[tbase + i];
    if (threadIdx.x < CK) wsv[threadIdx.x] = wsb[cb * CK + threadIdx.x];
    __syncthreads();

    for (int i = threadIdx.x; i < CT * CK; i += 256) {  // out += rwi@state
      int t = i >> 6, kv = i & 63;
      float acc = 0.f;
#pragma unroll 8
      for (int kk = 0; kk < CK; ++kk) acc += rs[t][kk] * st[kk][kv];
      outp[tbase + i] += acc;
    }
    __syncthreads();

    for (int i = threadIdx.x; i < CK * CK; i += 256) {  // state update
      int kk = i >> 6, kv = i & 63;
      st[kk][kv] = st[kk][kv] * wsv[kk] + wkvb[cb * CK * CK + i];
    }
    __syncthreads();
  }
  for (int i = threadIdx.x; i < CK * CK; i += 256)
    state_out[(size_t)bh * CK * CK + i] = st[i >> 6][i & 63];
}

// ---------------------------------------------------------------------------
// groupnorm (per b,t,h over K=64) * gn_w + gn_b, times silu gate -> bf16
// ---------------------------------------------------------------------------
__global__ __launch_bounds__(64)
void gnorm(const float* __restrict__ outp, const float* __restrict__ g,
           const float* __restrict__ gnw, const float* __restrict__ gnb,
           __bf16* __restrict__ y, int Hh, int T, int C) {
  __shared__ float red[64];
  int gid = blockIdx.x;
  int h = gid % Hh;
  int bt = gid / Hh;
  int b = bt / T, t = bt - b * T;
  size_t src = (((size_t)b * Hh + h) * T + t) * 64;
  int kk = threadIdx.x;
  float v = outp[src + kk];
  red[kk] = v;
  __syncthreads();
  for (int off = 32; off > 0; off >>= 1) {
    if (kk < off) red[kk] += red[kk + off];
    __syncthreads();
  }
  float mu = red[0] * (1.f / 64.f);
  __syncthreads();
  float dv = v - mu;
  red[kk] = dv * dv;
  __syncthreads();
  for (int off = 32; off > 0; off >>= 1) {
    if (kk < off) red[kk] += red[kk + off];
    __syncthreads();
  }
  float var = red[0] * (1.f / 64.f);
  float nv = dv * rsqrtf(var + 6.4e-4f);  // EPS = 1e-5*64
  int c = h * 64 + kk;
  size_t di = (size_t)bt * C + c;
  y[di] = f2bf((nv * gnw[c] + gnb[c]) * g[di]);
}

__global__ __launch_bounds__(256)
void lasttok(const float* __restrict__ x, float* __restrict__ dst, int T, int C) {
  int i = blockIdx.x * 256 + threadIdx.x;
  int b = i / C, c = i - b * C;
  dst[i] = x[((size_t)b * T + (T - 1)) * C + c];
}

// ---------------------------------------------------------------------------
extern "C" void kernel_launch(void* const* d_in, const int* in_sizes, int n_in,
                              void* d_out, int out_size, void* d_ws,
                              size_t ws_size, hipStream_t stream) {
  enum { Bb = 4, Tt = 2048, Cc = 2048, Hh = 32, Kk = 64, Nn = 64 };
  const int BT = Bb * Tt;                       // 8192
  const size_t BTC = (size_t)BT * Cc;           // 16,777,216

  const float* x    = (const float*)d_in[0];
  const float* shft = (const float*)d_in[1];
  const float* wkvs = (const float*)d_in[2];
  const float* tmx  = (const float*)d_in[3];
  const float* tmw  = (const float*)d_in[4];
  const float* tmk  = (const float*)d_in[5];
  const float* tmv  = (const float*)d_in[6];
  const float* tmr  = (const float*)d_in[7];
  const float* tmg  = (const float*)d_in[8];
  const float* w1   = (const float*)d_in[9];    // (C,160)
  const float* w2   = (const float*)d_in[10];   // (5,32,C)
  const float* td   = (const float*)d_in[11];
  const float* tdw1 = (const float*)d_in[12];   // (C,64)
  const float* tdw2 = (const float*)d_in[13];   // (64,C)
  const float* u    = (const float*)d_in[14];
  const float* Wr   = (const float*)d_in[15];
  const float* Wk   = (const float*)d_in[16];
  const float* Wv   = (const float*)d_in[17];
  const float* Wg   = (const float*)d_in[18];
  const float* Wo   = (const float*)d_in[19];
  const float* gnw  = (const float*)d_in[20];
  const float* gnb  = (const float*)d_in[21];

  float* out = (float*)d_out;                   // [BTC | B*C | B*H*K*K]
  float* shift_out = out + BTC;
  float* state_out = shift_out + (size_t)Bb * Cc;

  char* w = (char*)d_ws;
  size_t off = 0;
  auto alloc = [&](size_t bytes) -> void* {
    void* p = w + off;
    off = (off + bytes + 255) & ~(size_t)255;
    return p;
  };
  float*  dxf   = (float*)alloc(BTC * 4);
  __bf16* xmixb = (__bf16*)alloc(BTC * 2);
  float*  xxxf  = (float*)alloc((size_t)BT * 160 * 4);
  __bf16* xkb   = (__bf16*)alloc(BTC * 2);
  __bf16* xvb   = (__bf16*)alloc(BTC * 2);
  __bf16* xrb   = (__bf16*)alloc(BTC * 2);
  __bf16* xgb   = (__bf16*)alloc(BTC * 2);
  __bf16* xwb   = (__bf16*)alloc(BTC * 2);
  float*  rbf   = (float*)alloc(BTC * 4);
  float*  kbf   = (float*)alloc(BTC * 4);
  float*  vbf   = (float*)alloc(BTC * 4);
  float*  gf    = (float*)alloc(BTC * 4);
  float*  wtf   = (float*)alloc((size_t)BT * 64 * 4);
  float*  wlogf = (float*)alloc(BTC * 4);
  float*  outp  = (float*)alloc(BTC * 4);
  float*  rwif  = (float*)alloc(BTC * 4);
  float*  wkvb  = (float*)alloc((size_t)Bb * Hh * Nn * 64 * 64 * 4);
  float*  wsb   = (float*)alloc((size_t)Bb * Hh * Nn * 64 * 4);
  __bf16* yb    = (__bf16*)alloc(BTC * 2);
  __bf16* W1t   = (__bf16*)alloc((size_t)160 * Cc * 2);
  __bf16* td1t  = (__bf16*)alloc((size_t)64 * Cc * 2);
  __bf16* Wrb   = (__bf16*)alloc((size_t)Cc * Cc * 2);
  __bf16* Wkb   = (__bf16*)alloc((size_t)Cc * Cc * 2);
  __bf16* Wvb   = (__bf16*)alloc((size_t)Cc * Cc * 2);
  __bf16* Wgb   = (__bf16*)alloc((size_t)Cc * Cc * 2);
  __bf16* Wob   = (__bf16*)alloc((size_t)Cc * Cc * 2);

  // 1. premix
  premix<<<(int)(BTC / 256), 256, 0, stream>>>(x, shft, tmx, dxf, xmixb, Tt, Cc);

  // 2. weight conversions
  int nW = Cc * Cc;
  cvt_bf16_t<<<(Cc * 160 + 255) / 256, 256, 0, stream>>>(w1, W1t, Cc, 160);
  cvt_bf16_t<<<(Cc * 64 + 255) / 256, 256, 0, stream>>>(tdw1, td1t, Cc, 64);
  cvt_bf16<<<(nW + 255) / 256, 256, 0, stream>>>(Wr, Wrb, nW);
  cvt_bf16<<<(nW + 255) / 256, 256, 0, stream>>>(Wk, Wkb, nW);
  cvt_bf16<<<(nW + 255) / 256, 256, 0, stream>>>(Wv, Wvb, nW);
  cvt_bf16<<<(nW + 255) / 256, 256, 0, stream>>>(Wg, Wgb, nW);
  cvt_bf16<<<(nW + 255) / 256, 256, 0, stream>>>(Wo, Wob, nW);

  // 3. xxx = tanh(xmix @ W1)   (M=8192, N=160, K=2048)
  gemm_bf16<1><<<dim3(3, BT / 128), 256, 0, stream>>>(xmixb, W1t, xxxf, BT, 160,
                                                      Cc, Hh, Tt);
  // 4. m-einsum + lerps
  mix_lerp<<<BT, 256, 0, stream>>>(x, dxf, xxxf, w2, tmw, tmk, tmv, tmr, tmg,
                                   xkb, xvb, xrb, xgb, xwb, Cc);
  // 5. r,k,v projections scattered to (B,H,T,64); g with silu
  dim3 gBig(Cc / 64, BT / 128);
  gemm_bf16<2><<<gBig, 256, 0, stream>>>(xrb, Wrb, rbf, BT, Cc, Cc, Hh, Tt);
  gemm_bf16<2><<<gBig, 256, 0, stream>>>(xkb, Wkb, kbf, BT, Cc, Cc, Hh, Tt);
  gemm_bf16<2><<<gBig, 256, 0, stream>>>(xvb, Wvb, vbf, BT, Cc, Cc, Hh, Tt);
  gemm_bf16<3><<<gBig, 256, 0, stream>>>(xgb, Wgb, gf, BT, Cc, Cc, Hh, Tt);
  // 6. decay MLP stage 1: tanh(xw @ td_w1)  (N=64)
  gemm_bf16<1><<<dim3(1, BT / 128), 256, 0, stream>>>(xwb, td1t, wtf, BT, 64,
                                                      Cc, Hh, Tt);
  // 7. decay stage 2 + exp/clip/log -> wlog (B,H,T,64)
  decay<<<BT, 256, 0, stream>>>(wtf, tdw2, td, wlogf, Cc, Tt, Hh);
  // 8. per-chunk work
  chunk_kernel<<<Bb * Hh * Nn, 256, 0, stream>>>(rbf, kbf, vbf, wlogf, u, outp,
                                                 wkvb, wsb, rwif, Hh, Tt, Nn);
  // 9. sequential state scan per (b,h); writes final state to d_out tail
  scan_kernel<<<Bb * Hh, 256, 0, stream>>>(wkvs, wsb, wkvb, rwif, outp,
                                           state_out, Tt, Nn);
  // 10. groupnorm * gate -> bf16
  gnorm<<<BT * Hh, 64, 0, stream>>>(outp, gf, gnw, gnb, yb, Hh, Tt, Cc);
  // 11. final projection straight into d_out
  gemm_bf16<0><<<gBig, 256, 0, stream>>>(yb, Wob, out, BT, Cc, Cc, Hh, Tt);
  // 12. shift-state output = x[:, -1, :]
  lasttok<<<(Bb * Cc) / 256, 256, 0, stream>>>(x, shift_out, Tt, Cc);

  (void)in_sizes; (void)n_in; (void)out_size; (void)ws_size;
}
